// RigidDocking_28604482191509
// MI455X (gfx1250) — matrix operand order repaired and checked by reading, not compiled, not run on hardware
//
#include <hip/hip_runtime.h>
#include <hip/hip_fp16.h>

// ---------------------------------------------------------------------------
// RigidDocking SAKE network for MI455X (gfx1250, wave32, WMMA).
// All heavy (n^2-row) dense layers go through v_wmma_f32_16x16x32_f16 with
// f32 accumulation and fused epilogues (bias / scale / silu / tanh / residual).
// B operands are pre-swizzled into WMMA fragment order and zero-padded to a
// multiple of 64 columns, so the GEMM inner loop is branch-free:
// 10x global_load_b128 + 4x v_wmma per K-step.
// ---------------------------------------------------------------------------

typedef __attribute__((ext_vector_type(16))) _Float16 v16h;
typedef __attribute__((ext_vector_type(8)))  _Float16 v8h;
typedef __attribute__((ext_vector_type(8)))  float    v8f;

#define PI_F 3.14159265358979f

enum { ACT_NONE = 0, ACT_SILU = 1, ACT_TANH = 2 };

// ---------------------------------------------------------------------------
// WMMA GEMM: C(M,Nout) = act(scale*A(M,K)@B(K,Nout) + bias) [+ resid]
// A: f16 row-major (lda). B: f16 fragment order (pack_bfrag), padded to
// Npad = ceil64(Nout) columns; NT = Npad/16:
//   Bf[((kt*NT + nt)*32 + lane)*16 + e],  K = kt*32 + (lane>>4)*16 + e,
//                                         N = nt*16 + (lane&15)
// One wave computes a 16(M) x 64(N) strip with 4 accumulators (branch-free
// K loop); stores are guarded per tile by Nout. M tiles of 16 (wave-uniform
// guard); K multiple of 32.
// ---------------------------------------------------------------------------
__global__ __launch_bounds__(128)
void gemm_wmma_kernel(const _Float16* __restrict__ A, int lda,
                      const _Float16* __restrict__ Bf, int NT,
                      const float* __restrict__ bias, int biasN,
                      float* __restrict__ Cf, _Float16* __restrict__ Ch, int ldc,
                      const float* __restrict__ resid,
                      int M, int Nout, int K, float scale, int act)
{
    const int lane = threadIdx.x & 31;
    const int wave = threadIdx.x >> 5;
    const int m0 = (blockIdx.y * 4 + wave) << 4;
    if (m0 >= M) return;                       // wave-uniform: EXEC stays all-ones
    const int r  = lane & 15;
    const int hi = lane >> 4;                  // half-wave select
    const int nbase = blockIdx.x << 6;
    const int nt0 = blockIdx.x << 2;

    v8f acc0 = {0.f,0.f,0.f,0.f,0.f,0.f,0.f,0.f};
    v8f acc1 = acc0, acc2 = acc0, acc3 = acc0;

    const _Float16* Arow  = A + (size_t)(m0 + r) * lda + hi * 16;
    const _Float16* Bbase = Bf + ((size_t)nt0 * 32 + lane) * 16;
    const size_t bstep = (size_t)NT * 512;     // halves per K-tile in Bf

    for (int k0 = 0, kt = 0; k0 < K; k0 += 32, ++kt) {
        // A fragment (16x32): elems 0..15 -> K=k0+hi*16+e (matches B pack).
        // Two aligned 16B loads.
        v8h alo = *(const v8h*)(Arow + k0);
        v8h ahi = *(const v8h*)(Arow + k0 + 8);
        v16h a;
        #pragma unroll
        for (int e = 0; e < 8; ++e) { a[e] = alo[e]; a[e + 8] = ahi[e]; }

        const _Float16* Bp = Bbase + (size_t)kt * bstep;
        if (k0 + 32 < K) {
            __builtin_prefetch(Arow + k0 + 32, 0, 1);          // global_prefetch_b8
            __builtin_prefetch(Bp + bstep, 0, 1);
        }
        v8h b0lo = *(const v8h*)(Bp);
        v8h b0hi = *(const v8h*)(Bp + 8);
        v8h b1lo = *(const v8h*)(Bp + 512);
        v8h b1hi = *(const v8h*)(Bp + 520);
        v8h b2lo = *(const v8h*)(Bp + 1024);
        v8h b2hi = *(const v8h*)(Bp + 1032);
        v8h b3lo = *(const v8h*)(Bp + 1536);
        v8h b3hi = *(const v8h*)(Bp + 1544);
        v16h b0, b1, b2, b3;
        #pragma unroll
        for (int e = 0; e < 8; ++e) {
            b0[e] = b0lo[e]; b0[e + 8] = b0hi[e];
            b1[e] = b1lo[e]; b1[e + 8] = b1hi[e];
            b2[e] = b2lo[e]; b2[e + 8] = b2hi[e];
            b3[e] = b3lo[e]; b3[e + 8] = b3hi[e];
        }
        acc0 = __builtin_amdgcn_wmma_f32_16x16x32_f16(false, a, false, b0,
                                                      (short)0, acc0, false, false);
        acc1 = __builtin_amdgcn_wmma_f32_16x16x32_f16(false, a, false, b1,
                                                      (short)0, acc1, false, false);
        acc2 = __builtin_amdgcn_wmma_f32_16x16x32_f16(false, a, false, b2,
                                                      (short)0, acc2, false, false);
        acc3 = __builtin_amdgcn_wmma_f32_16x16x32_f16(false, a, false, b3,
                                                      (short)0, acc3, false, false);
    }

    auto emit = [&](const v8f& acc, int ncol0) {
        const int col = ncol0 + r;
        const float bv = (bias && col < biasN) ? bias[col] : 0.f;
        #pragma unroll
        for (int v = 0; v < 8; ++v) {
            const int row = m0 + v + hi * 8;   // C/D layout: VGPR v -> row v + 8*hi
            float val = acc[v] * scale + bv;
            if (act == ACT_SILU)      val = val / (1.f + expf(-val));
            else if (act == ACT_TANH) val = tanhf(val);
            const size_t o = (size_t)row * ldc + col;
            if (resid) val += resid[o];
            if (Cf) Cf[o] = val;
            if (Ch) Ch[o] = (_Float16)val;
        }
    };
    emit(acc0, nbase);
    if (nbase + 16 < Nout) emit(acc1, nbase + 16);
    if (nbase + 32 < Nout) emit(acc2, nbase + 32);
    if (nbase + 48 < Nout) emit(acc3, nbase + 48);
}

// ---------------------------------------------------------------------------
// f32 -> f16 row-major pack with zero padding: dst(dr,dc) from src(sr,sc)
// ---------------------------------------------------------------------------
__global__ void pack_f16_kernel(const float* __restrict__ src, int sr, int sc,
                                _Float16* __restrict__ dst, int dr, int dc)
{
    int id = blockIdx.x * blockDim.x + threadIdx.x;
    if (id >= dr * dc) return;
    int r = id / dc, c = id % dc;
    float v = (r < sr && c < sc) ? src[(size_t)r * sc + c] : 0.f;
    dst[id] = (_Float16)v;
}

// ---------------------------------------------------------------------------
// f32 -> f16 B-operand pack into WMMA fragment order (zero-padded to
// Kpad x Npad). transpose=0: B[k][n]=src[k][n]; transpose=1: B[k][n]=src[n][k].
// ---------------------------------------------------------------------------
__global__ void pack_bfrag_kernel(const float* __restrict__ src, int sr, int sc,
                                  int transpose, _Float16* __restrict__ dst,
                                  int Kpad, int Npad)
{
    int id = blockIdx.x * blockDim.x + threadIdx.x;
    int NT = Npad >> 4;
    int total = (Kpad >> 5) * NT * 512;
    if (id >= total) return;
    int e    = id & 15;
    int lane = (id >> 4) & 31;
    int fid  = id >> 9;                 // kt*NT + nt
    int kt = fid / NT, nt = fid - kt * NT;
    int K = kt * 32 + (lane >> 4) * 16 + e;
    int N = nt * 16 + (lane & 15);
    float v = 0.f;
    if (!transpose) { if (K < sr && N < sc) v = src[(size_t)K * sc + N]; }
    else            { if (N < sr && K < sc) v = src[(size_t)N * sc + K]; }
    dst[id] = (_Float16)v;
}

// ---------------------------------------------------------------------------
// Pairwise geometry + hh = [h[j], h[i]] in f16
// ---------------------------------------------------------------------------
__global__ void pair_setup_kernel(const float* __restrict__ h, const float* __restrict__ x,
                                  int n, float* __restrict__ norm, float* __restrict__ dirs,
                                  _Float16* __restrict__ hh)
{
    int i = blockIdx.x, j = threadIdx.x;
    if (j >= n) return;
    float d0 = x[j*3+0] - x[i*3+0];
    float d1 = x[j*3+1] - x[i*3+1];
    float d2 = x[j*3+2] - x[i*3+2];
    float sq = d0*d0 + d1*d1 + d2*d2;
    float nm = sqrtf(fmaxf(sq, 0.f) + 1e-10f);
    int p = i * n + j;
    norm[p] = nm;
    float inv = 1.f / (nm + 1e-5f);
    dirs[(size_t)p*3+0] = d0 * inv;
    dirs[(size_t)p*3+1] = d1 * inv;
    dirs[(size_t)p*3+2] = d2 * inv;
    size_t b = (size_t)p * 128;
    for (int c = 0; c < 64; ++c) {
        hh[b + c]      = (_Float16)h[(size_t)j*64 + c];
        hh[b + 64 + c] = (_Float16)h[(size_t)i*64 + c];
    }
}

// ---------------------------------------------------------------------------
// Edge feature: [hh(128), rbf*hk(50), norm(1), pad(13)] -> f16 (ld 192)
// ---------------------------------------------------------------------------
__global__ void edge_feat_kernel(const float* __restrict__ norm, const float* __restrict__ hk,
                                 const _Float16* __restrict__ hh,
                                 const float* __restrict__ betas, const float* __restrict__ means,
                                 _Float16* __restrict__ e16)
{
    int p = blockIdx.x, c = threadIdx.x;   // c < 192
    size_t o = (size_t)p * 192 + c;
    if (c < 128) { e16[o] = hh[(size_t)p * 128 + c]; return; }
    float nm = norm[p];
    if (c < 178) {
        int f = c - 128;
        float cut = (nm < 5.f) ? 0.5f * (cosf(nm * PI_F / 5.f) + 1.f) : 0.f;
        float dlt = expf(-nm) - means[f];
        float rbf = cut * expf(-betas[f] * dlt * dlt);
        e16[o] = (_Float16)(rbf * hk[(size_t)p * 64 + f]);
    } else if (c == 178) {
        e16[o] = (_Float16)nm;
    } else {
        e16[o] = (_Float16)0.f;
    }
}

// ---------------------------------------------------------------------------
// Combined euclidean x semantic attention, normalized over j. One thread per
// (node i, head). sem logits ld 16, att ld 4.
// ---------------------------------------------------------------------------
__device__ inline float celu2(float x) {
    return (x > 0.f) ? x : 2.f * (expf(0.5f * x) - 1.f);
}

__global__ void attention_kernel(const float* __restrict__ norm, const float* __restrict__ sem,
                                 int n, float* __restrict__ att)
{
    int id = blockIdx.x * blockDim.x + threadIdx.x;
    if (id >= n * 4) return;
    int i = id >> 2, hd = id & 3;
    const float* nr = norm + (size_t)i * n;
    const float* sr = sem + (size_t)i * n * 16;
    float me = -1e30f, ms = -1e30f;
    for (int j = 0; j < n; ++j) {
        float msk = (j == i) ? 1e5f : 0.f;
        me = fmaxf(me, -(nr[j] + msk));
        ms = fmaxf(ms, celu2(sr[(size_t)j*16 + hd]) - msk);
    }
    float se = 0.f, ss = 0.f;
    for (int j = 0; j < n; ++j) {
        float msk = (j == i) ? 1e5f : 0.f;
        se += expf(-(nr[j] + msk) - me);
        ss += expf(celu2(sr[(size_t)j*16 + hd]) - msk - ms);
    }
    float tot = 0.f;
    for (int j = 0; j < n; ++j) {
        float msk = (j == i) ? 1e5f : 0.f;
        float eu = expf(-(nr[j] + msk) - me) / se;
        float sm = expf(celu2(sr[(size_t)j*16 + hd]) - msk - ms) / ss;
        float a = eu * sm;
        att[((size_t)i * n + j) * 4 + hd] = a;
        tot += a;
    }
    float inv = 1.f / tot;
    for (int j = 0; j < n; ++j) att[((size_t)i * n + j) * 4 + hd] *= inv;
}

// h_e_att[p, f*4+hd] = e[p,f] * att[p,hd]  (f16, ld 256)
__global__ void mix_kernel(const float* __restrict__ e, const float* __restrict__ att,
                           _Float16* __restrict__ hea)
{
    int p = blockIdx.x, c = threadIdx.x;   // c < 256
    hea[(size_t)p * 256 + c] =
        (_Float16)(e[(size_t)p * 64 + (c >> 2)] * att[(size_t)p * 4 + (c & 3)]);
}

// h_e[i,c] = sum_j hea[i*n+j, c]
__global__ void sumj_kernel(const _Float16* __restrict__ hea, int n, float* __restrict__ he)
{
    int id = blockIdx.x * blockDim.x + threadIdx.x;
    if (id >= n * 256) return;
    int i = id >> 8, c = id & 255;
    const _Float16* p = hea + (size_t)i * n * 256 + c;
    float s = 0.f;
    for (int j = 0; j < n; ++j) s += (float)p[(size_t)j * 256];
    he[id] = s;
}

// comb[i,c,:] = (1/n) sum_j coeff[i*n+j,c] * dirs[i*n+j,:]; cnorm = |comb|^2
__global__ void einsum_kernel(const float* __restrict__ coeff, const float* __restrict__ dirs,
                              int n, _Float16* __restrict__ cn16)
{
    int id = blockIdx.x * blockDim.x + threadIdx.x;
    if (id >= n * 256) return;
    int i = id >> 8, c = id & 255;
    const float* cp = coeff + (size_t)i * n * 256 + c;
    const float* dp = dirs + (size_t)i * n * 3;
    float a0 = 0.f, a1 = 0.f, a2 = 0.f;
    for (int j = 0; j < n; ++j) {
        float cf = cp[(size_t)j * 256];
        a0 += cf * dp[j*3+0];
        a1 += cf * dp[j*3+1];
        a2 += cf * dp[j*3+2];
    }
    float inv = 1.f / (float)n;
    a0 *= inv; a1 *= inv; a2 *= inv;
    cn16[id] = (_Float16)(a0*a0 + a1*a1 + a2*a2);
}

// upd = [h(64), h_e(256), hc(64)] f16
__global__ void concat_kernel(const float* __restrict__ h, const float* __restrict__ he,
                              const float* __restrict__ hc, _Float16* __restrict__ upd)
{
    int i = blockIdx.x, c = threadIdx.x;   // c < 384
    float v = (c < 64) ? h[(size_t)i*64 + c]
            : (c < 320) ? he[(size_t)i*256 + (c - 64)]
            : hc[(size_t)i*64 + (c - 320)];
    upd[(size_t)i * 384 + c] = (_Float16)v;
}

// ---------------------------------------------------------------------------
// Cross-graph attention softmaxes (dot: 192 x 176)
// ---------------------------------------------------------------------------
__global__ void softmax_rows_kernel(const float* __restrict__ dot, _Float16* __restrict__ a016)
{
    int i = threadIdx.x;
    if (i >= 192) return;
    const float* row = dot + (size_t)i * 176;
    float m = -1e30f;
    for (int j = 0; j < 176; ++j) m = fmaxf(m, row[j]);
    float s = 0.f;
    for (int j = 0; j < 176; ++j) s += expf(row[j] - m);
    float inv = 1.f / s;
    for (int j = 0; j < 176; ++j) a016[(size_t)i*192 + j] = (_Float16)(expf(row[j] - m) * inv);
    for (int j = 176; j < 192; ++j) a016[(size_t)i*192 + j] = (_Float16)0.f;  // K-pad
}

__global__ void softmax_cols_kernel(const float* __restrict__ dot, _Float16* __restrict__ a1T16)
{
    int j = threadIdx.x;
    if (j >= 176) return;
    float m = -1e30f;
    for (int i = 0; i < 192; ++i) m = fmaxf(m, dot[(size_t)i*176 + j]);
    float s = 0.f;
    for (int i = 0; i < 192; ++i) s += expf(dot[(size_t)i*176 + j] - m);
    float inv = 1.f / s;
    for (int i = 0; i < 192; ++i)
        a1T16[(size_t)j*192 + i] = (_Float16)(expf(dot[(size_t)i*176 + j] - m) * inv);
}

// cat = [h, h - g] f16 (row softmax sums to 1 -> m = h - attn@other)
__global__ void cat_kernel(const float* __restrict__ h, const float* __restrict__ g,
                           _Float16* __restrict__ cat)
{
    int i = blockIdx.x, c = threadIdx.x;  // c < 128
    float v = (c < 64) ? h[(size_t)i*64 + c]
                       : (h[(size_t)i*64 + (c - 64)] - g[(size_t)i*64 + (c - 64)]);
    cat[(size_t)i * 128 + c] = (_Float16)v;
}

// y[k,d] = (1/n) sum_j x[j,d] * c[j,k]   (c ld 16)
__global__ void keypoints_kernel(const float* __restrict__ x, const float* __restrict__ c,
                                 int n, float* __restrict__ y)
{
    int t = threadIdx.x;
    if (t >= 9) return;
    int k = t / 3, d = t % 3;
    float s = 0.f;
    for (int j = 0; j < n; ++j) s += x[(size_t)j*3 + d] * c[(size_t)j*16 + k];
    y[k*3 + d] = s / (float)n;
}

// ---------------------------------------------------------------------------
// Single-thread 3x3 Kabsch (Jacobi eigensolver on A^T A), literal per source.
// ---------------------------------------------------------------------------
__device__ inline float det3(const float M[3][3]) {
    return M[0][0]*(M[1][1]*M[2][2]-M[1][2]*M[2][1])
         - M[0][1]*(M[1][0]*M[2][2]-M[1][2]*M[2][0])
         + M[0][2]*(M[1][0]*M[2][1]-M[1][1]*M[2][0]);
}

__global__ void kabsch_kernel(const float* __restrict__ y0, const float* __restrict__ y1,
                              float* __restrict__ out)
{
    if (threadIdx.x != 0 || blockIdx.x != 0) return;
    float A[3][3], S[3][3], V[3][3], U[3][3];
    for (int p = 0; p < 3; ++p)
        for (int q = 0; q < 3; ++q)
            A[p][q] = y1[p*3+0]*y0[q*3+0] + y1[p*3+1]*y0[q*3+1] + y1[p*3+2]*y0[q*3+2];
    for (int i = 0; i < 3; ++i)
        for (int j = 0; j < 3; ++j) {
            float s = 0.f;
            for (int k = 0; k < 3; ++k) s += A[k][i] * A[k][j];
            S[i][j] = s;
            V[i][j] = (i == j) ? 1.f : 0.f;
        }
    for (int sweep = 0; sweep < 30; ++sweep) {
        for (int t = 0; t < 3; ++t) {
            int p = (t == 2) ? 1 : 0;
            int q = (t == 0) ? 1 : 2;
            float apq = S[p][q];
            if (fabsf(apq) < 1e-12f) continue;
            float tau = (S[q][q] - S[p][p]) / (2.f * apq);
            float tt = ((tau >= 0.f) ? 1.f : -1.f) / (fabsf(tau) + sqrtf(1.f + tau*tau));
            float c = 1.f / sqrtf(1.f + tt*tt), s = tt * c;
            for (int k = 0; k < 3; ++k) { float a=S[k][p], b=S[k][q]; S[k][p]=c*a-s*b; S[k][q]=s*a+c*b; }
            for (int k = 0; k < 3; ++k) { float a=S[p][k], b=S[q][k]; S[p][k]=c*a-s*b; S[q][k]=s*a+c*b; }
            for (int k = 0; k < 3; ++k) { float a=V[k][p], b=V[k][q]; V[k][p]=c*a-s*b; V[k][q]=s*a+c*b; }
        }
    }
    float ev[3] = { S[0][0], S[1][1], S[2][2] };
    for (int a = 0; a < 2; ++a)
        for (int b = a + 1; b < 3; ++b)
            if (ev[b] > ev[a]) {
                float t = ev[a]; ev[a] = ev[b]; ev[b] = t;
                for (int k = 0; k < 3; ++k) { float tv = V[k][a]; V[k][a] = V[k][b]; V[k][b] = tv; }
            }
    float sig2 = 0.f;
    for (int k = 0; k < 3; ++k) {
        float u0 = 0.f, u1 = 0.f, u2 = 0.f;
        for (int m = 0; m < 3; ++m) { u0 += A[0][m]*V[m][k]; u1 += A[1][m]*V[m][k]; u2 += A[2][m]*V[m][k]; }
        float sg = sqrtf(fmaxf(u0*u0 + u1*u1 + u2*u2, 0.f));
        if (k == 2) sig2 = sg;
        if (sg > 1e-8f) { U[0][k]=u0/sg; U[1][k]=u1/sg; U[2][k]=u2/sg; }
        else            { U[0][k]=0.f;   U[1][k]=0.f;   U[2][k]=0.f;   }
    }
    if (sig2 <= 1e-8f) {
        U[0][2] = U[1][0]*U[2][1] - U[2][0]*U[1][1];
        U[1][2] = U[2][0]*U[0][1] - U[0][0]*U[2][1];
        U[2][2] = U[0][0]*U[1][1] - U[1][0]*U[0][1];
    }
    float d = (det3(U) * det3(V) >= 0.f) ? 1.f : -1.f;
    float sk[3] = { 1.f, 1.f, d };
    float R[3][3];
    for (int i = 0; i < 3; ++i)
        for (int j = 0; j < 3; ++j) {
            float acc = 0.f;
            for (int k = 0; k < 3; ++k) acc += U[i][k] * sk[k] * V[k][j];  // u2 @ s @ u1.T
            R[i][j] = acc;
        }
    float y0m[3], y1m[3];
    for (int dd = 0; dd < 3; ++dd) {
        y0m[dd] = (y0[dd] + y0[3+dd] + y0[6+dd]) / 3.f;
        y1m[dd] = (y1[dd] + y1[3+dd] + y1[6+dd]) / 3.f;
    }
    for (int i = 0; i < 3; ++i)
        for (int j = 0; j < 3; ++j) {
            float acc = 0.f;
            for (int m = 0; m < 3; ++m) acc += y0[i*3+m] * R[m][j];
            out[i*3+j] = acc + (y1m[j] - R[i][j] * y0m[j]);     // t = y1m - r*y0m
        }
    for (int k = 0; k < 9; ++k) out[9 + k] = y1[k];
}

// ---------------------------------------------------------------------------
// Host driver
// ---------------------------------------------------------------------------
static inline void gemm(hipStream_t st, const _Float16* A, int lda, const _Float16* Bf,
                        const float* bias, int biasN, float* Cf, _Float16* Ch, int ldc,
                        const float* resid, int M, int Nout, int K, float scale, int act)
{
    const int Npad = (Nout + 63) & ~63;        // B is packed/padded to this width
    dim3 g(Npad / 64, (M + 63) / 64);
    gemm_wmma_kernel<<<g, 128, 0, st>>>(A, lda, Bf, Npad / 16, bias, biasN, Cf, Ch, ldc, resid,
                                        M, Nout, K, scale, act);
}

extern "C" void kernel_launch(void* const* d_in, const int* in_sizes, int n_in,
                              void* d_out, int out_size, void* d_ws, size_t ws_size,
                              hipStream_t stream)
{
    (void)in_sizes; (void)n_in; (void)out_size; (void)ws_size;

    const float* h0_in = (const float*)d_in[0];   // (192,16)
    const float* x0    = (const float*)d_in[1];   // (192,3)
    const float* h1_in = (const float*)d_in[2];   // (176,16)
    const float* x1    = (const float*)d_in[3];   // (176,3)

    // params pytree leaves: sorted-key DFS (b before w)
    int pi = 4;
    auto P = [&]() { return (const float*)d_in[pi++]; };
    const float* emb_b = P(); const float* emb_w = P();     // emb_in
    const float* eo1_b = P(); const float* eo1_w = P();     // emb_out1
    const float* eo2_b = P(); const float* eo2_w = P();     // emb_out2
    const float* fc_b  = P(); const float* fc_w  = P();     // fc_combine
    const float* kp_w  = P();                               // fc_kp (w only)
    struct Layer {
        const float *betas, *means, *mlpin_b, *mlpin_w, *out1_b, *out1_w, *out2_b, *out2_w,
                    *node1_b, *node1_w, *node2_b, *node2_w, *post1_b, *post1_w, *post2_b,
                    *post2_w, *sem_b, *sem_w, *xmix_w;
        _Float16 *Wmlpin, *Wout1, *Wout2, *Wsem, *Wxmix, *Wpost1, *Wpost2, *Wnode1, *Wnode2;
    } Ls[4];
    for (int l = 0; l < 4; ++l) {
        Layer& L = Ls[l];
        L.betas = P(); L.means = P();
        L.mlpin_b = P(); L.mlpin_w = P();
        L.out1_b = P(); L.out1_w = P();
        L.out2_b = P(); L.out2_w = P();
        L.node1_b = P(); L.node1_w = P();
        L.node2_b = P(); L.node2_w = P();
        L.post1_b = P(); L.post1_w = P();
        L.post2_b = P(); L.post2_w = P();
        L.sem_b = P(); L.sem_w = P();
        L.xmix_w = P();
    }

    // workspace bump allocator (256B aligned)
    char* base = (char*)d_ws;
    size_t off = 0;
    auto alloc = [&](size_t bytes) -> void* {
        off = (off + 255) & ~(size_t)255;
        void* p = base + off;
        off += bytes;
        return p;
    };
    auto fA = [&](size_t e) { return (float*)alloc(e * 4); };
    auto hA = [&](size_t e) { return (_Float16*)alloc(e * 2); };

    float* h0 = fA(192 * 64);
    float* h1 = fA(192 * 64);
    // fragment-packed B weights (N padded to multiple of 64)
    _Float16* Wemb = hA(32 * 64);
    _Float16* Wfc  = hA(128 * 64);
    _Float16* Weo1 = hA(64 * 64);
    _Float16* Weo2 = hA(64 * 64);
    _Float16* Wkp  = hA(64 * 64);
    for (int l = 0; l < 4; ++l) {
        Ls[l].Wmlpin = hA(128 * 64);
        Ls[l].Wout1  = hA(192 * 64);
        Ls[l].Wout2  = hA(64 * 64);
        Ls[l].Wsem   = hA(64 * 64);
        Ls[l].Wxmix  = hA(256 * 256);
        Ls[l].Wpost1 = hA(256 * 64);
        Ls[l].Wpost2 = hA(64 * 64);
        Ls[l].Wnode1 = hA(384 * 64);
        Ls[l].Wnode2 = hA(64 * 64);
    }
    _Float16* hin0 = hA(192 * 32);
    _Float16* hin1 = hA(176 * 32);
    _Float16* h016  = hA(192 * 64);   // A-side row-major pack of h0 (or h)
    _Float16* h0Bf  = hA(192 * 64);   // B-frag pack of h0 (Kpad 192)
    _Float16* h1Bf  = hA(192 * 64);   // B-frag pack of h1 (Kpad 192, zero rows)
    _Float16* h1TBf = hA(64 * 192);   // B-frag pack of h1^T (64 x 176 -> Npad 192)
    float* dot = fA(192 * 176);
    _Float16* a016  = hA(192 * 192);
    _Float16* a1T16 = hA(176 * 192);
    float* g0 = fA(192 * 64);
    float* g1 = fA(192 * 64);
    _Float16* cat0 = hA(192 * 128);
    _Float16* cat1 = hA(192 * 128);
    _Float16* t16  = hA(192 * 64);
    _Float16* ho16 = hA(192 * 64);
    float* cbuf = fA(192 * 16);
    float* y0b = fA(16);
    float* y1b = fA(16);

    // pairwise scratch (sized for n=192). coeff overlays hh/hk/ein (dead then).
    const size_t P2 = 192 * 192;
    float* Sreg = fA(P2 * 256);                               // 37.75 MB union
    _Float16* hh16  = (_Float16*)Sreg;                        // P2*128 f16
    float*    hk    = (float*)((char*)Sreg + P2 * 128 * 2);   // P2*64 f32
    _Float16* ein16 = (_Float16*)((char*)Sreg + P2 * 128 * 2 + P2 * 64 * 4);  // P2*192 f16
    float* coeff = Sreg;                                      // P2*256 f32
    float* normb = fA(P2);
    float* dirs  = fA(P2 * 3);
    _Float16* mid16 = hA(P2 * 64);
    float* ef32 = fA(P2 * 64);
    _Float16* ef16 = hA(P2 * 64);
    float* semb = fA(P2 * 16);
    float* attb = fA(P2 * 4);
    _Float16* hea16 = hA(P2 * 256);
    float* h_e = fA(192 * 256);
    _Float16* cn16 = hA(192 * 256);
    float* hcb = fA(192 * 64);
    _Float16* p1_16 = hA(192 * 64);
    _Float16* upd16 = hA(192 * 384);
    _Float16* nd16  = hA(192 * 64);

    auto pk = [&](const float* s, int sr, int sc, _Float16* d, int dr, int dc) {
        int tot = dr * dc;
        pack_f16_kernel<<<(tot + 255) / 256, 256, 0, stream>>>(s, sr, sc, d, dr, dc);
    };
    auto pbf = [&](const float* s, int sr, int sc, int tr, _Float16* d, int Kpad, int Nout) {
        int Npad = (Nout + 63) & ~63;
        int tot = (Kpad / 32) * (Npad / 16) * 512;
        pack_bfrag_kernel<<<(tot + 255) / 256, 256, 0, stream>>>(s, sr, sc, tr, d, Kpad, Npad);
    };

    // pack weights into fragment order (zero-padded) once per launch
    pbf(emb_w, 16, 64, 0, Wemb, 32, 64);
    pbf(fc_w, 128, 64, 0, Wfc, 128, 64);
    pbf(eo1_w, 64, 64, 0, Weo1, 64, 64);
    pbf(eo2_w, 64, 64, 0, Weo2, 64, 64);
    pbf(kp_w, 64, 3, 0, Wkp, 64, 16);
    for (int l = 0; l < 4; ++l) {
        pbf(Ls[l].mlpin_w, 128, 50, 0, Ls[l].Wmlpin, 128, 64);
        pbf(Ls[l].out1_w, 179, 64, 0, Ls[l].Wout1, 192, 64);
        pbf(Ls[l].out2_w, 64, 64, 0, Ls[l].Wout2, 64, 64);
        pbf(Ls[l].sem_w, 64, 4, 0, Ls[l].Wsem, 64, 16);
        pbf(Ls[l].xmix_w, 256, 256, 0, Ls[l].Wxmix, 256, 256);
        pbf(Ls[l].post1_w, 256, 64, 0, Ls[l].Wpost1, 256, 64);
        pbf(Ls[l].post2_w, 64, 64, 0, Ls[l].Wpost2, 64, 64);
        pbf(Ls[l].node1_w, 384, 64, 0, Ls[l].Wnode1, 384, 64);
        pbf(Ls[l].node2_w, 64, 64, 0, Ls[l].Wnode2, 64, 64);
    }
    pk(h0_in, 192, 16, hin0, 192, 32);
    pk(h1_in, 176, 16, hin1, 176, 32);

    // input embedding
    gemm(stream, hin0, 32, Wemb, emb_b, 64, h0, nullptr, 64, nullptr, 192, 64, 32, 1.f, ACT_NONE);
    gemm(stream, hin1, 32, Wemb, emb_b, 64, h1, nullptr, 64, nullptr, 176, 64, 32, 1.f, ACT_NONE);

    auto sake = [&](const Layer& L, float* h, const float* x, int n) {
        int n2 = n * n;
        pair_setup_kernel<<<n, 256, 0, stream>>>(h, x, n, normb, dirs, hh16);
        gemm(stream, hh16, 128, L.Wmlpin, L.mlpin_b, 50, hk, nullptr, 64, nullptr,
             n2, 64, 128, 1.f, ACT_NONE);
        edge_feat_kernel<<<n2, 192, 0, stream>>>(normb, hk, hh16, L.betas, L.means, ein16);
        gemm(stream, ein16, 192, L.Wout1, L.out1_b, 64, nullptr, mid16, 64, nullptr,
             n2, 64, 192, 1.f, ACT_SILU);
        gemm(stream, mid16, 64, L.Wout2, L.out2_b, 64, ef32, ef16, 64, nullptr,
             n2, 64, 64, 1.f, ACT_NONE);
        gemm(stream, ef16, 64, L.Wsem, L.sem_b, 4, semb, nullptr, 16, nullptr,
             n2, 16, 64, 1.f, ACT_NONE);
        attention_kernel<<<(n * 4 + 255) / 256, 256, 0, stream>>>(normb, semb, n, attb);
        mix_kernel<<<n2, 256, 0, stream>>>(ef32, attb, hea16);
        sumj_kernel<<<(n * 256 + 255) / 256, 256, 0, stream>>>(hea16, n, h_e);
        gemm(stream, hea16, 256, L.Wxmix, nullptr, 0, coeff, nullptr, 256, nullptr,
             n2, 256, 256, 1.f, ACT_TANH);
        einsum_kernel<<<(n * 256 + 255) / 256, 256, 0, stream>>>(coeff, dirs, n, cn16);
        gemm(stream, cn16, 256, L.Wpost1, L.post1_b, 64, nullptr, p1_16, 64, nullptr,
             n, 64, 256, 1.f, ACT_SILU);
        gemm(stream, p1_16, 64, L.Wpost2, L.post2_b, 64, hcb, nullptr, 64, nullptr,
             n, 64, 64, 1.f, ACT_SILU);
        concat_kernel<<<n, 384, 0, stream>>>(h, h_e, hcb, upd16);
        gemm(stream, upd16, 384, L.Wnode1, L.node1_b, 64, nullptr, nd16, 64, nullptr,
             n, 64, 384, 1.f, ACT_SILU);
        gemm(stream, nd16, 64, L.Wnode2, L.node2_b, 64, h, nullptr, 64, h,
             n, 64, 64, 1.f, ACT_SILU);   // h += silu(...)
    };

    auto cross = [&]() {
        pk(h0, 192, 64, h016, 192, 64);            // A of dot
        pbf(h0, 192, 64, 0, h0Bf, 192, 64);        // B of g1
        pbf(h1, 176, 64, 0, h1Bf, 192, 64);        // B of g0 (zero K-pad rows)
        pbf(h1, 176, 64, 1, h1TBf, 64, 176);       // B of dot = h1^T (Npad 192)
        gemm(stream, h016, 64, h1TBf, nullptr, 0, dot, nullptr, 176, nullptr,
             192, 176, 64, 0.125f, ACT_NONE);
        softmax_rows_kernel<<<1, 192, 0, stream>>>(dot, a016);
        softmax_cols_kernel<<<1, 192, 0, stream>>>(dot, a1T16);
        gemm(stream, a016, 192, h1Bf, nullptr, 0, g0, nullptr, 64, nullptr,
             192, 64, 192, 1.f, ACT_NONE);
        gemm(stream, a1T16, 192, h0Bf, nullptr, 0, g1, nullptr, 64, nullptr,
             176, 64, 192, 1.f, ACT_NONE);
        cat_kernel<<<192, 128, 0, stream>>>(h0, g0, cat0);
        cat_kernel<<<176, 128, 0, stream>>>(h1, g1, cat1);
        gemm(stream, cat0, 128, Wfc, fc_b, 64, h0, nullptr, 64, nullptr,
             192, 64, 128, 1.f, ACT_SILU);
        gemm(stream, cat1, 128, Wfc, fc_b, 64, h1, nullptr, 64, nullptr,
             176, 64, 128, 1.f, ACT_SILU);
    };

    for (int l = 0; l < 4; ++l) {
        sake(Ls[l], h0, x0, 192);
        sake(Ls[l], h1, x1, 176);
        cross();
    }

    auto head = [&](float* h, const float* x, int n, float* y) {
        pk(h, n, 64, h016, n, 64);
        gemm(stream, h016, 64, Weo1, eo1_b, 64, nullptr, t16, 64, nullptr,
             n, 64, 64, 1.f, ACT_SILU);
        gemm(stream, t16, 64, Weo2, eo2_b, 64, nullptr, ho16, 64, nullptr,
             n, 64, 64, 1.f, ACT_NONE);
        gemm(stream, ho16, 64, Wkp, nullptr, 0, cbuf, nullptr, 16, nullptr,
             n, 16, 64, 1.f, ACT_NONE);
        keypoints_kernel<<<1, 16, 0, stream>>>(x, cbuf, n, y);
    };
    head(h0, x0, 192, y0b);
    head(h1, x1, 176, y1b);

    kabsch_kernel<<<1, 1, 0, stream>>>(y0b, y1b, (float*)d_out);
}
